// DotProductLabelling_22368189678011
// MI455X (gfx1250) — matrix operand order repaired and checked by reading, not compiled, not run on hardware
//
#include <hip/hip_runtime.h>
#include <hip/hip_bf16.h>

#define NNODE 131072
#define NEDGE 262144
#define CCH   256
#define SDIM  5
#define BN_EPS 1e-5f

typedef __attribute__((ext_vector_type(16))) __bf16 v16bf;
typedef __attribute__((ext_vector_type(8)))  __bf16 v8bf;
typedef __attribute__((ext_vector_type(8)))  float  v8f;
typedef __attribute__((ext_vector_type(4)))  float  v4f;

// stats layout (floats): [0:256) feat sum, [256:512) feat sumsq,
//                        [512:517) struct sum, [517:522) struct sumsq
// coef  layout (floats): [0:256) feat scale, [256:512) feat shift,
//                        [512:517) struct scale, [517:522) struct shift

__global__ void k_init(const float* __restrict__ Wf, __bf16* __restrict__ Wb,
                       float* __restrict__ stats) {
    int t = blockIdx.x * 256 + threadIdx.x;
    if (t < CCH * CCH) Wb[t] = (__bf16)Wf[t];
    if (t < 522) stats[t] = 0.0f;
}

__global__ void k_struct(const float* __restrict__ propped,
                         const float* __restrict__ Ws,
                         const float* __restrict__ bs,
                         float* __restrict__ z, float* __restrict__ stats) {
    __shared__ float red[8];
    int e = blockIdx.x * 256 + threadIdx.x;
    float p[SDIM], zz[SDIM];
#pragma unroll
    for (int i = 0; i < SDIM; i++) p[i] = propped[(size_t)e * SDIM + i];
#pragma unroll
    for (int j = 0; j < SDIM; j++) {
        float acc = bs[j];
#pragma unroll
        for (int i = 0; i < SDIM; i++) acc += p[i] * Ws[j * SDIM + i];
        zz[j] = acc;
        z[(size_t)e * SDIM + j] = acc;
    }
    int lane = threadIdx.x & 31, wave = threadIdx.x >> 5;
#pragma unroll
    for (int j = 0; j < SDIM; j++) {
        for (int pass = 0; pass < 2; ++pass) {
            float v = pass ? zz[j] * zz[j] : zz[j];
            for (int off = 16; off; off >>= 1) v += __shfl_xor(v, off, 32);
            if (lane == 0) red[wave] = v;
            __syncthreads();
            if (threadIdx.x == 0) {
                float s = 0.f;
#pragma unroll
                for (int w = 0; w < 8; w++) s += red[w];
                atomicAdd(&stats[512 + pass * SDIM + j], s);
            }
            __syncthreads();
        }
    }
}

// Block tile: 64 edges x 256 cols. 8 waves: wm in {0,1} (32 rows each),
// wn in {0..3} (64 cols each). Per wave: 2 A-frags x 4 B-frags = 8 WMMAs/K-step,
// so each B fragment (L2-resident W) feeds two WMMAs.
__global__ void __launch_bounds__(256, 2)
k_gemm(const float* __restrict__ x, const int* __restrict__ edges,
       const __bf16* __restrict__ Wb, float* __restrict__ h,
       float* __restrict__ stats) {
    __shared__ __bf16 lA[64 * 264];   // 64 edge rows x 256 K, +8 pad -> no bank conflicts
    const int tid = threadIdx.x;
    const int eb  = blockIdx.x * 64;

    // Stage A tile: bf16(x[src] * x[dst]) for 64 edges x 256 channels.
    // (b_feat dropped: constant per-column bias cancels exactly in BN.)
#pragma unroll
    for (int c = 0; c < 8; c++) {
        int idx = tid + c * 256;            // 2048 chunks of 8 floats
        int r   = idx >> 5;                 // edge row 0..63
        int col = (idx & 31) << 3;          // 0,8,...,248
        int e0 = edges[eb + r];
        int e1 = edges[NEDGE + eb + r];
        const v4f* p0 = (const v4f*)(x + (size_t)e0 * CCH + col);
        const v4f* p1 = (const v4f*)(x + (size_t)e1 * CCH + col);
        v4f a0 = p0[0], a1 = p0[1];
        v4f b0 = p1[0], b1 = p1[1];
        v8bf o;
#pragma unroll
        for (int i = 0; i < 4; i++) {
            o[i]     = (__bf16)(a0[i] * b0[i]);
            o[4 + i] = (__bf16)(a1[i] * b1[i]);
        }
        *(v8bf*)&lA[r * 264 + col] = o;
    }
    __syncthreads();

    const int wave = tid >> 5, lane = tid & 31;
    const int wm = wave & 1;        // M tile: 32 rows
    const int wn = wave >> 1;       // N tile: 64 cols
    const int half = lane >> 4, l15 = lane & 15;
    const int arow0 = wm * 32 + l15;        // first 16-row sub-tile
    const int arow1 = arow0 + 16;           // second 16-row sub-tile

    v8f acc[2][4] = {};
#pragma unroll
    for (int k0 = 0; k0 < CCH; k0 += 32) {
        // A fragments: contiguous K order (k = 16*half + s), matching B
        v8bf a0lo = *(const v8bf*)&lA[arow0 * 264 + k0 + half * 16];
        v8bf a0hi = *(const v8bf*)&lA[arow0 * 264 + k0 + 8 + half * 16];
        v16bf a0 = __builtin_shufflevector(a0lo, a0hi,
            0, 1, 2, 3, 4, 5, 6, 7, 8, 9, 10, 11, 12, 13, 14, 15);
        v8bf a1lo = *(const v8bf*)&lA[arow1 * 264 + k0 + half * 16];
        v8bf a1hi = *(const v8bf*)&lA[arow1 * 264 + k0 + 8 + half * 16];
        v16bf a1 = __builtin_shufflevector(a1lo, a1hi,
            0, 1, 2, 3, 4, 5, 6, 7, 8, 9, 10, 11, 12, 13, 14, 15);
        // B fragments: column n = W_feat row n (contiguous K), L2-resident
        const __bf16* bp = Wb + (size_t)(wn * 64 + l15) * CCH + k0 + half * 16;
        v16bf b0 = *(const v16bf*)(bp);
        v16bf b1 = *(const v16bf*)(bp + 16 * CCH);
        v16bf b2 = *(const v16bf*)(bp + 32 * CCH);
        v16bf b3 = *(const v16bf*)(bp + 48 * CCH);
        acc[0][0] = __builtin_amdgcn_wmma_f32_16x16x32_bf16(false, a0, false, b0, (short)0, acc[0][0], false, false);
        acc[1][0] = __builtin_amdgcn_wmma_f32_16x16x32_bf16(false, a1, false, b0, (short)0, acc[1][0], false, false);
        acc[0][1] = __builtin_amdgcn_wmma_f32_16x16x32_bf16(false, a0, false, b1, (short)0, acc[0][1], false, false);
        acc[1][1] = __builtin_amdgcn_wmma_f32_16x16x32_bf16(false, a1, false, b1, (short)0, acc[1][1], false, false);
        acc[0][2] = __builtin_amdgcn_wmma_f32_16x16x32_bf16(false, a0, false, b2, (short)0, acc[0][2], false, false);
        acc[1][2] = __builtin_amdgcn_wmma_f32_16x16x32_bf16(false, a1, false, b2, (short)0, acc[1][2], false, false);
        acc[0][3] = __builtin_amdgcn_wmma_f32_16x16x32_bf16(false, a0, false, b3, (short)0, acc[0][3], false, false);
        acc[1][3] = __builtin_amdgcn_wmma_f32_16x16x32_bf16(false, a1, false, b3, (short)0, acc[1][3], false, false);
    }

    // Store pre-BN h and accumulate per-column batch stats.
#pragma unroll
    for (int t = 0; t < 4; t++) {
        int n = wn * 64 + t * 16 + l15;
        float s1 = 0.f, s2 = 0.f;
#pragma unroll
        for (int mt = 0; mt < 2; mt++) {
            int mbase = eb + wm * 32 + mt * 16 + half * 8;
#pragma unroll
            for (int rr = 0; rr < 8; rr++) {
                float v = acc[mt][t][rr];
                h[(size_t)(mbase + rr) * CCH + n] = v;
                s1 += v; s2 += v * v;
            }
        }
        s1 += __shfl_xor(s1, 16, 32);
        s2 += __shfl_xor(s2, 16, 32);
        if (half == 0) {
            atomicAdd(&stats[n], s1);
            atomicAdd(&stats[CCH + n], s2);
        }
    }
}

__global__ void k_final(const float* __restrict__ g_feat, const float* __restrict__ beta_feat,
                        const float* __restrict__ g_s, const float* __restrict__ beta_s,
                        const float* __restrict__ stats, float* __restrict__ coef) {
    int t = threadIdx.x;
    const float invE = 1.0f / (float)NEDGE;
    if (t < CCH) {
        float mu  = stats[t] * invE;
        float var = stats[CCH + t] * invE - mu * mu;
        float sc  = g_feat[t] * rsqrtf(var + BN_EPS);
        coef[t]        = sc;
        coef[CCH + t]  = beta_feat[t] - mu * sc;
    }
    if (t < SDIM) {
        float mu  = stats[512 + t] * invE;
        float var = stats[517 + t] * invE - mu * mu;
        float sc  = g_s[t] * rsqrtf(var + BN_EPS);
        coef[512 + t] = sc;
        coef[517 + t] = beta_s[t] - mu * sc;
    }
}

__global__ void k_logit(const float* __restrict__ h, const float* __restrict__ z,
                        const float* __restrict__ coef, const float* __restrict__ Wcls,
                        const float* __restrict__ bcls, float* __restrict__ out) {
    __shared__ float sc[CCH], sh[CCH], wc[CCH];
    __shared__ float scoef[10], wcs[SDIM], sb;
    int tid = threadIdx.x;
    if (tid < CCH) { sc[tid] = coef[tid]; sh[tid] = coef[CCH + tid]; wc[tid] = Wcls[tid]; }
    if (tid < 10)   scoef[tid] = coef[512 + tid];
    if (tid < SDIM) wcs[tid] = Wcls[CCH + tid];
    if (tid == 0)   sb = bcls[0];
    __syncthreads();

    int wave = tid >> 5, lane = tid & 31;
    size_t e = (size_t)blockIdx.x * 8 + wave;
    const float* hr = h + e * CCH;
    float acc = 0.f;
#pragma unroll
    for (int i = 0; i < 8; i++) {
        int n = i * 32 + lane;                 // coalesced 128B per step
        float v = hr[n] * sc[n] + sh[n];
        v = v > 0.f ? v : 0.f;
        acc += v * wc[n];
    }
    for (int off = 16; off; off >>= 1) acc += __shfl_xor(acc, off, 32);
    if (lane == 0) {
        float st = 0.f;
#pragma unroll
        for (int j = 0; j < SDIM; j++) {
            float v = z[e * SDIM + j] * scoef[j] + scoef[5 + j];
            v = v > 0.f ? v : 0.f;
            st += v * wcs[j];
        }
        out[e] = acc + st + sb;
    }
}

extern "C" void kernel_launch(void* const* d_in, const int* in_sizes, int n_in,
                              void* d_out, int out_size, void* d_ws, size_t ws_size,
                              hipStream_t stream) {
    const float* x          = (const float*)d_in[0];
    const int*   edges      = (const int*)d_in[1];
    const float* propped    = (const float*)d_in[2];
    const float* W_feat     = (const float*)d_in[3];
    /* d_in[4] b_feat: cancels exactly under BatchNorm -> unused */
    const float* g_feat     = (const float*)d_in[5];
    const float* beta_feat  = (const float*)d_in[6];
    const float* W_struct   = (const float*)d_in[7];
    const float* b_struct   = (const float*)d_in[8];
    const float* g_struct   = (const float*)d_in[9];
    const float* beta_struct= (const float*)d_in[10];
    const float* W_cls      = (const float*)d_in[11];
    const float* b_cls      = (const float*)d_in[12];
    float* out = (float*)d_out;

    // workspace layout
    float*  h     = (float*)d_ws;                        // E*256 f32
    float*  z     = h + (size_t)NEDGE * CCH;             // E*5  f32
    __bf16* Wb    = (__bf16*)(z + (size_t)NEDGE * SDIM); // 256*256 bf16
    float*  stats = (float*)(Wb + CCH * CCH);            // 522 f32
    float*  coef  = stats + 522;                         // 522 f32

    k_init  <<<(CCH * CCH) / 256, 256, 0, stream>>>(W_feat, Wb, stats);
    k_struct<<<NEDGE / 256,       256, 0, stream>>>(propped, W_struct, b_struct, z, stats);
    k_gemm  <<<NEDGE / 64,        256, 0, stream>>>(x, edges, Wb, h, stats);
    k_final <<<1,                 256, 0, stream>>>(g_feat, beta_feat, g_struct, beta_struct, stats, coef);
    k_logit <<<NEDGE / 8,         256, 0, stream>>>(h, z, coef, W_cls, b_cls, out);
}
